// LSTMBlock_14362370638361
// MI455X (gfx1250) — matrix operand order repaired and checked by reading, not compiled, not run on hardware
//
#include <hip/hip_runtime.h>

typedef __attribute__((ext_vector_type(16))) _Float16 v16h;
typedef __attribute__((ext_vector_type(8)))  float    v8f;

#define BATCH   262144
#define HDIM    128
#define OUT_PLANE (BATCH * HDIM)   // 33554432

__device__ __forceinline__ unsigned pack_h2(float lo, float hi) {
    union { _Float16 h[2]; unsigned u; } p;
    p.h[0] = (_Float16)lo;
    p.h[1] = (_Float16)hi;
    return p.u;
}

__device__ __forceinline__ float fsig(float x) {
    return 1.0f / (1.0f + __expf(-x));
}
__device__ __forceinline__ float ftanh_fast(float x) {
    float e = __expf(2.0f * x);
    return (e - 1.0f) / (e + 1.0f);
}

// LDS: WMMA-B fragments of fused weights, f16.
// chunk id = (g*8 + t)*8 + k ; per chunk: 32 lanes x 8 dwords (1 KB).
// lane encodes (n = t*16 + lane%16, kbase = 16*(lane/16)); dword v packs K = kbase+2v (lo), kbase+2v+1 (hi).
__global__ __launch_bounds__(256)
void lstm_fused_wmma(const float* __restrict__ x,  const float* __restrict__ h,
                     const float* __restrict__ c,
                     const float* __restrict__ Wxi, const float* __restrict__ Whi, const float* __restrict__ bi,
                     const float* __restrict__ Wxf, const float* __restrict__ Whf, const float* __restrict__ bfv,
                     const float* __restrict__ Wxc, const float* __restrict__ Whc, const float* __restrict__ bc,
                     const float* __restrict__ Wxo, const float* __restrict__ Who, const float* __restrict__ bo,
                     float* __restrict__ out)
{
    __shared__ __align__(32) unsigned smem[65536];   // 256 KB

    const int tid = threadIdx.x;

    // ---- stage fused weights [K=256][N=512] as f16 WMMA B-fragments in LDS ----
    for (int it = tid; it < 65536; it += 256) {
        const int lane = it & 31;
        const int v    = (it >> 5) & 7;
        const int k    = (it >> 8) & 7;
        const int t    = (it >> 11) & 7;
        const int g    = (it >> 14) & 3;
        const float* wx = (g == 0) ? Wxi : (g == 1) ? Wxf : (g == 2) ? Wxc : Wxo;
        const float* wh = (g == 0) ? Whi : (g == 1) ? Whf : (g == 2) ? Whc : Who;
        const int n  = t * 16 + (lane & 15);
        const int kk = k * 32 + ((lane >> 4) << 4) + 2 * v;   // even, pair stays in same half
        const float* srcw = (kk < 128) ? (wx + kk * 128 + n) : (wh + (kk - 128) * 128 + n);
        smem[(((g * 8 + t) * 8 + k) * 32 + lane) * 8 + v] = pack_h2(srcw[0], srcw[128]);
    }
    __syncthreads();

    const int lane  = tid & 31;
    const int wave  = tid >> 5;                 // H-column tile 0..7
    const int col   = wave * 16 + (lane & 15);  // output column within H
    const int arow  = lane & 15;                // A-fragment row within tile
    const int khalf = (lane >> 4) << 4;         // 0 or 16: lane's K half
    const int row0b = blockIdx.x * 128;

    const float bias_i = bi[col];
    const float bias_f = bfv[col];
    const float bias_c = bc[col];
    const float bias_o = bo[col];

    const v16h* bfrag = (const v16h*)smem;      // one v16h per (chunk, lane)

    for (int m = 0; m < 8; ++m) {
        const int row0 = row0b + m * 16;
        v8f acc_i = {}, acc_f = {}, acc_c = {}, acc_o = {};

#pragma unroll
        for (int k = 0; k < 8; ++k) {
            // ---- A fragment: 16x32 f16 tile of [x|h] ----
            const float* src = (k < 4) ? x : h;
            const float* p = src + (size_t)(row0 + arow) * 128 + (k & 3) * 32 + khalf;
            float4 a0 = *(const float4*)(p);        // K = khalf+0..3
            float4 a1 = *(const float4*)(p + 4);    // K = khalf+4..7
            float4 a2 = *(const float4*)(p + 8);    // K = khalf+8..11
            float4 a3 = *(const float4*)(p + 12);   // K = khalf+12..15
            v16h af;
            af[0]  = (_Float16)a0.x; af[1]  = (_Float16)a0.y;
            af[2]  = (_Float16)a0.z; af[3]  = (_Float16)a0.w;
            af[4]  = (_Float16)a1.x; af[5]  = (_Float16)a1.y;
            af[6]  = (_Float16)a1.z; af[7]  = (_Float16)a1.w;
            af[8]  = (_Float16)a2.x; af[9]  = (_Float16)a2.y;
            af[10] = (_Float16)a2.z; af[11] = (_Float16)a2.w;
            af[12] = (_Float16)a3.x; af[13] = (_Float16)a3.y;
            af[14] = (_Float16)a3.z; af[15] = (_Float16)a3.w;

            // ---- B fragments for the 4 gates, this wave's column tile ----
            v16h b0 = bfrag[((0 * 8 + wave) * 8 + k) * 32 + lane];
            v16h b1 = bfrag[((1 * 8 + wave) * 8 + k) * 32 + lane];
            v16h b2 = bfrag[((2 * 8 + wave) * 8 + k) * 32 + lane];
            v16h b3 = bfrag[((3 * 8 + wave) * 8 + k) * 32 + lane];

            acc_i = __builtin_amdgcn_wmma_f32_16x16x32_f16(false, af, false, b0, (short)0, acc_i, false, false);
            acc_f = __builtin_amdgcn_wmma_f32_16x16x32_f16(false, af, false, b1, (short)0, acc_f, false, false);
            acc_c = __builtin_amdgcn_wmma_f32_16x16x32_f16(false, af, false, b2, (short)0, acc_c, false, false);
            acc_o = __builtin_amdgcn_wmma_f32_16x16x32_f16(false, af, false, b3, (short)0, acc_o, false, false);
        }

        // ---- epilogue: gates -> activations -> c/h update ----
        const int rbase = row0 + ((lane >> 4) << 3);   // +8 rows for upper half-wave
#pragma unroll
        for (int r = 0; r < 8; ++r) {
            const size_t idx = (size_t)(rbase + r) * 128 + col;
            const float iv = fsig(acc_i[r] + bias_i);
            const float fv = fsig(acc_f[r] + bias_f);
            const float gv = ftanh_fast(acc_c[r] + bias_c);
            const float ov = fsig(acc_o[r] + bias_o);
            const float cn = fv * c[idx] + iv * gv;
            const float hn = ov * ftanh_fast(cn);
            out[idx]                 = hn;   // h_new
            out[OUT_PLANE + idx]     = hn;   // h_new (tuple repeat)
            out[2 * OUT_PLANE + idx] = cn;   // c_new
        }
    }
}

extern "C" void kernel_launch(void* const* d_in, const int* in_sizes, int n_in,
                              void* d_out, int out_size, void* d_ws, size_t ws_size,
                              hipStream_t stream) {
    (void)in_sizes; (void)n_in; (void)out_size; (void)d_ws; (void)ws_size;
    const float* x   = (const float*)d_in[0];
    const float* h   = (const float*)d_in[1];
    const float* c   = (const float*)d_in[2];
    const float* Wxi = (const float*)d_in[3];
    const float* Whi = (const float*)d_in[4];
    const float* bi  = (const float*)d_in[5];
    const float* Wxf = (const float*)d_in[6];
    const float* Whf = (const float*)d_in[7];
    const float* bfv = (const float*)d_in[8];
    const float* Wxc = (const float*)d_in[9];
    const float* Whc = (const float*)d_in[10];
    const float* bc  = (const float*)d_in[11];
    const float* Wxo = (const float*)d_in[12];
    const float* Who = (const float*)d_in[13];
    const float* bo  = (const float*)d_in[14];
    float* out = (float*)d_out;

    dim3 grid(BATCH / 128);   // 2048 blocks, each covers 128 batch rows
    dim3 block(256);          // 8 waves
    lstm_fused_wmma<<<grid, block, 0, stream>>>(x, h, c,
                                                Wxi, Whi, bi,
                                                Wxf, Whf, bfv,
                                                Wxc, Whc, bc,
                                                Wxo, Who, bo,
                                                out);
}
